// AffineMorph_88983132439065
// MI455X (gfx1250) — matrix operand order repaired and checked by reading, not compiled, not run on hardware
//
#include <hip/hip_runtime.h>
#include <hip/hip_bf16.h>

// ---------------------------------------------------------------------------
// AffineMorph on MI455X (gfx1250).
//  * stride-2 3D convs as implicit GEMM on v_wmma_f32_16x16x32_f16 (wave32).
//    Gather offsets + validity flags packed in one LDS int per k (padded to a
//    multiple of 32 -> branchless, fully pipelined global gathers); weights
//    staged in LDS as f16 with rows padded to Kp (aligned ds_load_b128 reads).
//  * max-pool + FC head + 4x4 expm (scaling & squaring) on one tiny block.
//  * affine grid-pull (trilinear, dct2 reflect, zero outside) warp kernel.
// ---------------------------------------------------------------------------

typedef __attribute__((ext_vector_type(16))) _Float16 v16h;
typedef __attribute__((ext_vector_type(8)))  float    v8f;
typedef __attribute__((ext_vector_type(4)))  int      v4i;

#define MAX_KP  864            // ceil32(Cin*27) max (32*27 is already mult 32)
#define MAX_WH  (32 * 864)     // Cout*Kp max f16 weights staged in LDS

// Packed table bits: off[24:0] | dx2<<25 | dy2<<26 | dz2<<27 | sel<<28 | inv<<29
#define OFF_MASK 0x01FFFFFF
#define F_DX2    (1 << 25)
#define F_DY2    (1 << 26)
#define F_DZ2    (1 << 27)
#define F_SEL    (1 << 28)
#define F_INV    (1 << 29)

// ------------------------- conv3d stride-2 WMMA ----------------------------
// Implicit GEMM: M = output voxels (16/wave, 4 waves/block = 64 voxels/block),
// N = Cout (1 or 2 tiles of 16, A fragment reused), K = Cin*27 in steps of 32.
// 'SAME' padding for stride2/k3 on even sizes is (lo=0, hi=1): in = 2*out + d,
// so an access is OOB iff (out == outS-1 && tap == 2) in that dimension.
__global__ __launch_bounds__(128)
void conv3d_s2_wmma_kernel(const float* __restrict__ inA,
                           const float* __restrict__ inB, int chanA,
                           const float* __restrict__ w32,
                           const float* __restrict__ bias,
                           float* __restrict__ out,
                           int Cin, int Cout, int inS, int outS, int batch) {
  __shared__ __align__(16) int      s_tab[MAX_KP];
  __shared__ __align__(16) _Float16 s_w[MAX_WH];   // (Cout, Kp) row-major f16

  const int K   = Cin * 27;
  const int Kp  = (K + 31) & ~31;
  const int tid = threadIdx.x;

  // ---- build gather table (once per block; div/mod only here) ----
  for (int k = tid; k < Kp; k += 128) {
    int t;
    if (k < K) {
      int ci = k / 27;
      int r  = k - ci * 27;
      int dz = r / 9, dy = (r - dz * 9) / 3, dx = r - (r / 3) * 3;
      int sel = 0, cc = ci;
      if (ci >= chanA) { cc = ci - chanA; sel = F_SEL; }
      int off = ((cc * inS + dz) * inS + dy) * inS + dx;     // < 2^25 always
      t = off | ((dz == 2) ? F_DZ2 : 0) | ((dy == 2) ? F_DY2 : 0) |
          ((dx == 2) ? F_DX2 : 0) | sel;
    } else {
      t = F_INV;                                   // padded: invalid for all
    }
    s_tab[k] = t;
  }
  // ---- stage weights as f16 in LDS, rows padded/zeroed to Kp ----
  const int WN = Cout * Kp;
  for (int i = tid; i < WN; i += 128) {
    int co = i / Kp;
    int kk = i - co * Kp;
    s_w[i] = (kk < K) ? (_Float16)w32[co * K + kk] : (_Float16)0.f;
  }
  __syncthreads();

  const int lane = tid & 31;
  const int wv   = tid >> 5;
  const int S3o  = outS * outS * outS;
  const long long inS3 = (long long)inS * inS * inS;

  // block -> (batch, group of 4 M-tiles); S3o % 64 == 0 for every layer
  const int groups = S3o >> 6;
  const int b  = blockIdx.x / groups;
  const int mt = (blockIdx.x - b * groups) * 4 + wv;
  if (b >= batch) return;                          // uniform

  const int m     = lane & 15;
  const int voxel = mt * 16 + m;
  const int vx = voxel % outS;
  const int vy = (voxel / outS) % outS;
  const int vz = voxel / (outS * outS);

  const int sp = ((2 * vz) * inS + (2 * vy)) * inS + (2 * vx);
  const float* pA = inA + (long long)b * chanA * inS3 + sp;
  const float* pB = inB + (long long)b * (Cin - chanA) * inS3 + sp;
  // lane validity mask: a tap fails iff its flag bit is set here
  const int lm = ((vx == outS - 1) ? F_DX2 : 0) |
                 ((vy == outS - 1) ? F_DY2 : 0) |
                 ((vz == outS - 1) ? F_DZ2 : 0) | F_INV;

  __builtin_prefetch(pA, 0, 3);                    // global_prefetch_b8

  const int ntiles = Cout >> 4;                    // 1 or 2 (uniform)
  const int kA_hi  = (lane >= 16) ? 16 : 0;        // A: K 0-15 | 16-31
  const int kB_hi  = (lane >= 16) ? 16 : 0;        // B: K 0-15 | 16-31
  const int nrow   = lane & 15;
  const _Float16* wr0 = s_w + nrow * Kp;
  const _Float16* wr1 = s_w + (16 + nrow) * Kp;
  const v4i* tab4 = (const v4i*)s_tab;

  v8f acc0 = {}, acc1 = {};
  for (int ks = 0; ks < K; ks += 32) {
    // ---- 4 x ds_load_b128: this lane's 16 packed table entries ----
    union { v4i q[4]; int i[16]; } tt;
    const int qb = (ks + kA_hi) >> 2;              // 16B-aligned (kA_hi in {0,16})
    tt.q[0] = tab4[qb];
    tt.q[1] = tab4[qb + 1];
    tt.q[2] = tab4[qb + 2];
    tt.q[3] = tab4[qb + 3];

    // ---- branchless, pipelined A gather ----
    union { v16h v; _Float16 h[16]; } af;
#pragma unroll
    for (int e = 0; e < 16; ++e) {
      const int  t     = tt.i[e];
      const bool valid = (t & lm) == 0;
      const int  off   = valid ? (t & OFF_MASK) : 0;   // base is always legal
      const float* p   = (t & F_SEL) ? pB : pA;
      const float  raw = p[off];                        // unconditional load
      af.h[e] = valid ? (_Float16)raw : (_Float16)0.f;
    }

    // ---- B fragments: 2 x ds_load_b128 each (rows padded to Kp) ----
    union { v4i q[2]; v16h v; } bf0;
    {
      const v4i* w4 = (const v4i*)(wr0 + ks + kB_hi);
      bf0.q[0] = w4[0];
      bf0.q[1] = w4[1];
    }
    acc0 = __builtin_amdgcn_wmma_f32_16x16x32_f16(false, af.v, false, bf0.v,
                                                  (short)0, acc0, false, false);
    if (ntiles > 1) {
      union { v4i q[2]; v16h v; } bf1;
      const v4i* w4 = (const v4i*)(wr1 + ks + kB_hi);
      bf1.q[0] = w4[0];
      bf1.q[1] = w4[1];
      acc1 = __builtin_amdgcn_wmma_f32_16x16x32_f16(false, af.v, false, bf1.v,
                                                    (short)0, acc1, false,
                                                    false);
    }
  }

  // D layout: VGPR r, lanes 0-15 -> (M=r, N=lane); lanes 16-31 -> (M=r+8).
  const int mbase = (lane >= 16) ? 8 : 0;
  {
    union { v8f v; float f[8]; } u; u.v = acc0;
    const float bv = bias[nrow];
    float* op = out + ((long long)b * Cout + nrow) * S3o + mt * 16 + mbase;
#pragma unroll
    for (int r = 0; r < 8; ++r) {
      float x = u.f[r] + bv;
      op[r] = (x >= 0.f) ? x : 0.2f * x;
    }
  }
  if (ntiles > 1) {
    union { v8f v; float f[8]; } u; u.v = acc1;
    const float bv = bias[16 + nrow];
    float* op = out + ((long long)b * Cout + 16 + nrow) * S3o + mt * 16 + mbase;
#pragma unroll
    for (int r = 0; r < 8; ++r) {
      float x = u.f[r] + bv;
      op[r] = (x >= 0.f) ? x : 0.2f * x;
    }
  }
}

// ------------------------- global max pool ---------------------------------
__global__ __launch_bounds__(256)
void maxpool_kernel(const float* __restrict__ in, float* __restrict__ out,
                    int n) {
  __shared__ float red[256];
  const float* p = in + (long long)blockIdx.x * n;
  float m = -3.402823466e38f;
  for (int i = threadIdx.x; i < n; i += 256) m = fmaxf(m, p[i]);
  red[threadIdx.x] = m;
  __syncthreads();
  for (int s = 128; s > 0; s >>= 1) {
    if (threadIdx.x < s)
      red[threadIdx.x] = fmaxf(red[threadIdx.x], red[threadIdx.x + s]);
    __syncthreads();
  }
  if (threadIdx.x == 0) out[blockIdx.x] = red[0];
}

// ------------------------- FC head + expm + shift --------------------------
__device__ __forceinline__ void mm4(const float* A, const float* B, float* C) {
  for (int i = 0; i < 4; ++i)
    for (int j = 0; j < 4; ++j) {
      float s = 0.f;
      for (int k = 0; k < 4; ++k) s += A[i * 4 + k] * B[k * 4 + j];
      C[i * 4 + j] = s;
    }
}

__global__ __launch_bounds__(64)
void head_kernel(const float* __restrict__ pooled,
                 const float* __restrict__ wf0, const float* __restrict__ bf0,
                 const float* __restrict__ wf1, const float* __restrict__ bf1,
                 float* __restrict__ Amat /* per batch: 3x4 row-major */) {
  __shared__ float y[2][32];
  __shared__ float prm[2][7];
  const int t = threadIdx.x;
  const int b = t >> 5;
  const int c = t & 31;

  float acc = bf0[c];
  for (int i = 0; i < 32; ++i) acc += pooled[b * 32 + i] * wf0[c * 32 + i];
  y[b][c] = (acc >= 0.f) ? acc : 0.2f * acc;
  __syncthreads();
  if (c < 7) {
    float a2 = bf1[c];
    for (int i = 0; i < 32; ++i) a2 += y[b][i] * wf1[c * 32 + i];
    prm[b][c] = a2;
  }
  __syncthreads();

  if (c == 0) {
    const float s2 = 0.7071067811865476f, s3 = 0.5773502691896258f;
    float M[16];
    for (int i = 0; i < 16; ++i) M[i] = 0.f;
    M[3]  = prm[b][0];  M[7] = prm[b][1];  M[11] = prm[b][2];   // translations
    M[1]  =  prm[b][3] * s2;  M[4] = -prm[b][3] * s2;           // rot (0,1)
    M[2]  =  prm[b][4] * s2;  M[8] = -prm[b][4] * s2;           // rot (0,2)
    M[6]  =  prm[b][5] * s2;  M[9] = -prm[b][5] * s2;           // rot (1,2)
    M[0] += prm[b][6] * s3; M[5] += prm[b][6] * s3; M[10] += prm[b][6] * s3;

    // scaling & squaring with order-12 Horner Taylor
    float nrm = 0.f;
    for (int i = 0; i < 4; ++i) {
      float rs = 0.f;
      for (int j = 0; j < 4; ++j) rs += fabsf(M[i * 4 + j]);
      nrm = fmaxf(nrm, rs);
    }
    int s = 0;
    while (nrm > 0.5f && s < 30) { nrm *= 0.5f; ++s; }
    float scale = 1.f;
    for (int i = 0; i < s; ++i) scale *= 0.5f;
    for (int i = 0; i < 16; ++i) M[i] *= scale;

    float P[16], T[16];
    for (int i = 0; i < 16; ++i) P[i] = (i % 5 == 0) ? 1.f : 0.f;  // I
    for (int j = 12; j >= 1; --j) {
      mm4(M, P, T);
      const float inv = 1.f / (float)j;
      for (int i = 0; i < 16; ++i)
        P[i] = T[i] * inv + ((i % 5 == 0) ? 1.f : 0.f);
    }
    for (int i = 0; i < s; ++i) {
      mm4(P, P, T);
      for (int q = 0; q < 16; ++q) P[q] = T[q];
    }

    // fold shift (origin at FOV center, c = 192/2): t' = t + c - R@c
    const float cc = 96.f;
    for (int i = 0; i < 3; ++i) {
      for (int k = 0; k < 3; ++k) Amat[b * 12 + i * 4 + k] = P[i * 4 + k];
      float tr = P[i * 4 + 3] + cc -
                 (P[i * 4 + 0] + P[i * 4 + 1] + P[i * 4 + 2]) * cc;
      Amat[b * 12 + i * 4 + 3] = tr;
    }
  }
}

// --------------------------- affine grid pull ------------------------------
__device__ __forceinline__ int reflect_dct2(int i, int n) {
  int p = i % (2 * n);
  if (p < 0) p += 2 * n;
  return (p < n) ? p : (2 * n - 1 - p);
}

__global__ __launch_bounds__(256)
void gridpull_kernel(const float* __restrict__ src,
                     const float* __restrict__ Amat,
                     float* __restrict__ out, int S, long long total) {
  long long idx = (long long)blockIdx.x * 256 + threadIdx.x;
  if (idx >= total) return;
  const long long S3 = (long long)S * S * S;
  const int b = (int)(idx / S3);
  long long v = idx - (long long)b * S3;
  const int x = (int)(v % S);
  const int y = (int)((v / S) % S);
  const int z = (int)(v / ((long long)S * S));

  const float* A = Amat + b * 12;
  const float fz = (float)z, fy = (float)y, fx = (float)x;
  const float g0 = A[0] * fz + A[1] * fy + A[2]  * fx + A[3];
  const float g1 = A[4] * fz + A[5] * fy + A[6]  * fx + A[7];
  const float g2 = A[8] * fz + A[9] * fy + A[10] * fx + A[11];

  const float lim = (float)S - 0.5f;
  const bool inb = (g0 > -0.5f) && (g0 < lim) && (g1 > -0.5f) && (g1 < lim) &&
                   (g2 > -0.5f) && (g2 < lim);

  const float f0 = floorf(g0), f1 = floorf(g1), f2 = floorf(g2);
  const float d0 = g0 - f0, d1 = g1 - f1, d2 = g2 - f2;
  const int i0 = (int)f0, i1 = (int)f1, i2 = (int)f2;

  const float* sp = src + (long long)b * S3;
  float outv = 0.f;
#pragma unroll
  for (int dz = 0; dz < 2; ++dz) {
    const int zz = reflect_dct2(i0 + dz, S);
    const float wz = dz ? d0 : 1.f - d0;
#pragma unroll
    for (int dy = 0; dy < 2; ++dy) {
      const int yy = reflect_dct2(i1 + dy, S);
      const float wy = dy ? d1 : 1.f - d1;
#pragma unroll
      for (int dx = 0; dx < 2; ++dx) {
        const int xx = reflect_dct2(i2 + dx, S);
        const float wx = dx ? d2 : 1.f - d2;
        outv += sp[((long long)zz * S + yy) * S + xx] * (wz * wy * wx);
      }
    }
  }
  out[idx] = inb ? outv : 0.f;
}

// ------------------------------- launcher ----------------------------------
extern "C" void kernel_launch(void* const* d_in, const int* in_sizes, int n_in,
                              void* d_out, int out_size, void* d_ws,
                              size_t ws_size, hipStream_t stream) {
  const float* src = (const float*)d_in[0];
  const float* tgt = (const float*)d_in[1];
  const float* we[4] = {(const float*)d_in[2], (const float*)d_in[4],
                        (const float*)d_in[6], (const float*)d_in[8]};
  const float* be[4] = {(const float*)d_in[3], (const float*)d_in[5],
                        (const float*)d_in[7], (const float*)d_in[9]};
  const float* wf0 = (const float*)d_in[10];
  const float* bf0 = (const float*)d_in[11];
  const float* wf1 = (const float*)d_in[12];
  const float* bf1 = (const float*)d_in[13];
  float* out = (float*)d_out;

  const int Cins[4]  = {2, 16, 32, 32};
  const int Couts[4] = {16, 32, 32, 32};
  const int inSs[4]  = {192, 96, 48, 24};
  const int outSs[4] = {96, 48, 24, 12};

  char* ws = (char*)d_ws;
  size_t off = 0;
  auto carve = [&](size_t bytes) -> char* {
    char* p = ws + off;
    off += (bytes + 255) & ~(size_t)255;
    return p;
  };
  float* act[4];
  for (int l = 0; l < 4; ++l)
    act[l] = (float*)carve((size_t)2 * Couts[l] * outSs[l] * outSs[l] *
                           outSs[l] * sizeof(float));
  float* pooled = (float*)carve(64 * sizeof(float));
  float* Amat   = (float*)carve(2 * 12 * sizeof(float));
  (void)ws_size; (void)in_sizes; (void)n_in; (void)out_size;

  // 1) conv stack (layer 0 consumes source|target concat via two pointers)
  for (int l = 0; l < 4; ++l) {
    const int S3o = outSs[l] * outSs[l] * outSs[l];
    const unsigned blocks = (unsigned)(2 * (S3o >> 6));  // 4 M-tiles / block
    const float* inA = (l == 0) ? src : act[l - 1];
    const float* inB = (l == 0) ? tgt : act[l - 1];
    const int chanA  = (l == 0) ? 1 : Cins[l];
    conv3d_s2_wmma_kernel<<<blocks, 128, 0, stream>>>(
        inA, inB, chanA, we[l], be[l], act[l], Cins[l], Couts[l], inSs[l],
        outSs[l], 2);
  }

  // 2) global max pool over 12^3 per (b, c)
  maxpool_kernel<<<64, 256, 0, stream>>>(act[3], pooled, 12 * 12 * 12);

  // 3) FC head + matrix exponential + shift-affine fold
  head_kernel<<<1, 64, 0, stream>>>(pooled, wf0, bf0, wf1, bf1, Amat);

  // 4) affine grid pull (trilinear, dct2 reflect, zero outside FOV)
  const long long total = 2LL * 192 * 192 * 192;
  gridpull_kernel<<<(unsigned)((total + 255) / 256), 256, 0, stream>>>(
      src, Amat, out, 192, total);
}
